// SOGAE_88330297409968
// MI455X (gfx1250) — matrix operand order repaired and checked by reading, not compiled, not run on hardware
//
#include <hip/hip_runtime.h>

// ---------------------------------------------------------------------------
// Types (trivially-copyable ext-vector types, safe for bit_cast)
// ---------------------------------------------------------------------------
typedef __bf16         v16bf  __attribute__((ext_vector_type(16)));
typedef float          v8f    __attribute__((ext_vector_type(8)));
typedef unsigned int   u32x4  __attribute__((ext_vector_type(4)));
typedef int            i32x4  __attribute__((ext_vector_type(4)));
typedef int            i32x8  __attribute__((ext_vector_type(8)));
typedef float          f32x4  __attribute__((ext_vector_type(4)));
typedef unsigned short u16x8  __attribute__((ext_vector_type(8)));

struct FragBits { u32x4 lo, hi; };   // 32 bytes == v16bf

#define B_DIM   256
#define G_DIM   4096
#define E_DIM   64
#define E2_DIM  4096
#define KDIM    4096
#define NDIM    4096
#define KC_CNT  128            // K / 32
#define NT_CNT  256            // N / 16
#define MT_CNT  16             // M / 16
#define N_STEP  24
#define BETA    1e-3f

#if defined(__gfx1250__) && __has_builtin(__builtin_amdgcn_tensor_load_to_lds)
#define HAVE_TDM 1
#else
#define HAVE_TDM 0
#endif

// round-to-nearest-even f32 -> bf16 (raw ushort)
__device__ __forceinline__ unsigned short f2bf(float f) {
    unsigned int u = __builtin_bit_cast(unsigned int, f);
    u += 0x7FFFu + ((u >> 16) & 1u);
    return (unsigned short)(u >> 16);
}

// ---------------------------------------------------------------------------
// Packed operand layouts (per-lane WMMA fragment order):
//  A: pA[((mTile*128 + kc)*32 + lane)*16 + e]
//     lane: row = mTile*16 + (lane&15); elem e: k = kc*32 + (lane>>4)*8 + e + (e>=8 ? 8 : 0)
//  B: pB[((kc*256 + nt)*32 + lane)*16 + e]
//     lane: col = nt*16 + (lane&15);   elem e: k = kc*32 + (lane>>4)*8 + e + (e>=8 ? 8 : 0)
// ---------------------------------------------------------------------------

// ---- one-time weight pack: W = xw (TR=false) or W = xw^T (TR=true) --------
template <bool TR>
__global__ __launch_bounds__(256) void pack_B(const float* __restrict__ xw,
                                              unsigned short* __restrict__ pB) {
    __shared__ unsigned short tile[32][72];   // [k_local][n_local] (padded)
    const int t = threadIdx.x;
    const int ntBase = blockIdx.x * 4;        // 4 n-tiles = 64 cols
    const int kc = blockIdx.y;                // one 32-wide k-chunk
    const int n0 = ntBase * 16;
    const int k0 = kc * 32;

    if (!TR) {
        // W[k][n] = xw[k*4096 + n]: 32 rows(k) x 64 cols(n), coalesced rows
        int kl = t >> 3, c0 = (t & 7) * 8;
        const float* src = xw + (size_t)(k0 + kl) * 4096 + n0 + c0;
        f32x4 a = *(const f32x4*)(src);
        f32x4 b = *(const f32x4*)(src + 4);
        u16x8 o;
        o[0] = f2bf(a.x); o[1] = f2bf(a.y); o[2] = f2bf(a.z); o[3] = f2bf(a.w);
        o[4] = f2bf(b.x); o[5] = f2bf(b.y); o[6] = f2bf(b.z); o[7] = f2bf(b.w);
        *(u16x8*)(&tile[kl][c0]) = o;
    } else {
        // W[k][n] = xw[n*4096 + k]: 64 rows(n) x 32 cols(k), transpose on store
        int nl = t >> 2, c0 = (t & 3) * 8;
        const float* src = xw + (size_t)(n0 + nl) * 4096 + k0 + c0;
        f32x4 a = *(const f32x4*)(src);
        f32x4 b = *(const f32x4*)(src + 4);
        tile[c0 + 0][nl] = f2bf(a.x); tile[c0 + 1][nl] = f2bf(a.y);
        tile[c0 + 2][nl] = f2bf(a.z); tile[c0 + 3][nl] = f2bf(a.w);
        tile[c0 + 4][nl] = f2bf(b.x); tile[c0 + 5][nl] = f2bf(b.y);
        tile[c0 + 6][nl] = f2bf(b.z); tile[c0 + 7][nl] = f2bf(b.w);
    }
    __syncthreads();

    // emit packed fragments
    int f = t >> 1, h = t & 1;
    int ntL = f >> 5, lane = f & 31;
    int nl = ntL * 16 + (lane & 15);
    int klBase = (lane >> 4) * 8 + h * 16;
    u16x8 o;
#pragma unroll
    for (int i = 0; i < 8; ++i) o[i] = tile[klBase + i][nl];
    size_t dst = (((size_t)kc * NT_CNT + ntBase + ntL) * 32 + lane) * 16 + h * 8;
    *(u16x8*)(pB + dst) = o;
}

// ---- state init: y = 0, z[b,i] = z0[i] ------------------------------------
__global__ __launch_bounds__(256) void init_zy(float* __restrict__ y,
                                               float* __restrict__ z,
                                               const float* __restrict__ z0) {
    int idx = blockIdx.x * 256 + threadIdx.x;
    y[idx] = 0.0f;
    if (idx < B_DIM * E_DIM) z[idx] = z0[idx & (E_DIM - 1)];
}

// ---- A producer: pA = packed bf16(images - y) -----------------------------
__global__ __launch_bounds__(256) void sub_conv_packed(const float* __restrict__ images,
                                                       const float* __restrict__ y,
                                                       unsigned short* __restrict__ pA) {
    int t = blockIdx.x * 256 + threadIdx.x;   // 131072 threads
    int h = t & 1;
    int lane = (t >> 1) & 31;
    int kc = (t >> 6) & 127;
    int mTile = t >> 13;
    int row = mTile * 16 + (lane & 15);
    int k0 = kc * 32 + (lane >> 4) * 8 + h * 16;
    const float* ia = images + (size_t)row * 4096 + k0;
    const float* ya = y + (size_t)row * 4096 + k0;
    f32x4 a0 = *(const f32x4*)ia, a1 = *(const f32x4*)(ia + 4);
    f32x4 b0 = *(const f32x4*)ya, b1 = *(const f32x4*)(ya + 4);
    u16x8 o;
    o[0] = f2bf(a0.x - b0.x); o[1] = f2bf(a0.y - b0.y);
    o[2] = f2bf(a0.z - b0.z); o[3] = f2bf(a0.w - b0.w);
    o[4] = f2bf(a1.x - b1.x); o[5] = f2bf(a1.y - b1.y);
    o[6] = f2bf(a1.z - b1.z); o[7] = f2bf(a1.w - b1.w);
    *(u16x8*)(pA + (((size_t)mTile * KC_CNT + kc) * 32 + lane) * 16 + h * 8) = o;
}

// ---------------------------------------------------------------------------
// LDS stage loader: copies a (2 k-chunk x 8 n-tile) packed-B slab (16 KB).
// TDM path: one wave issues a 2-D tensor_load_to_lds (rows = 2 k-chunks of
// 8 KB, row stride 256 KB). Fallback: coalesced copy by all threads.
// ---------------------------------------------------------------------------
__device__ __forceinline__ void load_stage(unsigned short* ldsDst,
                                           const unsigned short* gsrc,
                                           int tid) {
#if HAVE_TDM
    if ((tid >> 5) == 0) {
        unsigned long long ga = (unsigned long long)(const void*)gsrc;
        unsigned lds = (unsigned)(unsigned long long)(void*)ldsDst;  // flat[31:0] = LDS addr
        const unsigned rowUnits = 1024u;      // 8192 B in 8-byte units
        const unsigned strideUnits = 32768u;  // 262144 B in 8-byte units
        u32x4 g0;
        g0[0] = 1u;                                   // count=1 (valid), load, user
        g0[1] = lds;                                  // lds_addr
        g0[2] = (unsigned)ga;                         // global_addr[31:0]
        g0[3] = (unsigned)((ga >> 32) & 0x01FFFFFFull) | (2u << 30);  // addr[56:32], type=2
        i32x8 g1;
        g1[0] = (int)(3u << 16);                                  // data_size=8B
        g1[1] = (int)((strideUnits & 0xFFFFu) << 16);             // tensor_dim0[15:0]
        g1[2] = (int)(((strideUnits >> 16) & 0xFFFFu) | (2u << 16)); // dim0[31:16] | dim1[15:0]=2
        g1[3] = (int)(rowUnits << 16);                            // dim1[31:16]=0 | tile_dim0
        g1[4] = 2;                                                // tile_dim1=2, tile_dim2=0
        g1[5] = (int)strideUnits;                                 // dim0_stride[31:0]
        g1[6] = 0;                                                // dim0_stride hi | dim1_stride lo
        g1[7] = 0;
        i32x4 zz4 = (i32x4)0;
        i32x8 zz8 = (i32x8)0;
        __builtin_amdgcn_tensor_load_to_lds(g0, g1, zz4, zz4, zz8, 0);
    }
#else
#pragma unroll
    for (int v = 0; v < 4; ++v) {
        int lin = tid + v * 256;              // 1024 chunks of 16 B
        int r = lin >> 9, c = lin & 511;
        u32x4 d = *(const u32x4*)(gsrc + (size_t)r * (NT_CNT * 512) + c * 8);
        *(u32x4*)(ldsDst + (r * 4096 + c * 8)) = d;
    }
#endif
}

// ---------------------------------------------------------------------------
// WMMA GEMM: C(256x4096 f32) (+)= A(256x4096 bf16 packed) @ W(4096x4096 bf16 packed)
// Block: 64x128 tile, 8 waves (2x4), each wave 32x32. Double-buffered B in LDS.
// ---------------------------------------------------------------------------
template <bool ACC>
__global__ __launch_bounds__(256) void gemm_wmma(const unsigned short* __restrict__ pA,
                                                 const unsigned short* __restrict__ pB,
                                                 float* __restrict__ C) {
    __shared__ unsigned short Blds[2][8192];  // 2 stages x (2 kc x 8 nt x 32 lanes x 16)
    const int tid  = threadIdx.x;
    const int lane = tid & 31;
    const int wave = tid >> 5;
    const int wm = wave >> 2, wn = wave & 3;
    const int l15 = lane & 15, rsel = lane >> 4;
    const int ntBase = blockIdx.x * 8;        // 8 n-tiles (128 cols)
    const int mtBase = blockIdx.y * 4;        // 4 m-tiles (64 rows)

    v8f acc[2][2] = {};

    const unsigned short* src0 = pB + ((size_t)0 * NT_CNT + ntBase) * 512;
    load_stage(&Blds[0][0], src0, tid);

    for (int kcp = 0; kcp < KC_CNT / 2; ++kcp) {
        const int s = kcp & 1;
        if (kcp + 1 < KC_CNT / 2) {
            const unsigned short* srcN =
                pB + (((size_t)(kcp + 1) * 2) * NT_CNT + ntBase) * 512;
            load_stage(&Blds[s ^ 1][0], srcN, tid);
        }
#if HAVE_TDM
        if (wave == 0) {
            if (kcp + 1 < KC_CNT / 2) __builtin_amdgcn_s_wait_tensorcnt(1);
            else                      __builtin_amdgcn_s_wait_tensorcnt(0);
        }
#endif
        __syncthreads();

#pragma unroll
        for (int kk = 0; kk < 2; ++kk) {
            const int kc = kcp * 2 + kk;
            v16bf af[2], bf_[2];
#pragma unroll
            for (int i = 0; i < 2; ++i) {
                const int mTile = mtBase + wm * 2 + i;
                const unsigned short* p =
                    pA + (((size_t)mTile * KC_CNT + kc) * 32 + lane) * 16;
                FragBits f;
                f.lo = *(const u32x4*)(p);
                f.hi = *(const u32x4*)(p + 8);
                af[i] = __builtin_bit_cast(v16bf, f);
            }
#pragma unroll
            for (int j = 0; j < 2; ++j) {
                const int ntL = wn * 2 + j;
                const unsigned short* q = &Blds[s][((kk * 8 + ntL) * 32 + lane) * 16];
                FragBits f;
                f.lo = *(const u32x4*)(q);
                f.hi = *(const u32x4*)(q + 8);
                bf_[j] = __builtin_bit_cast(v16bf, f);
            }
#pragma unroll
            for (int i = 0; i < 2; ++i)
#pragma unroll
                for (int j = 0; j < 2; ++j)
                    acc[i][j] = __builtin_amdgcn_wmma_f32_16x16x32_bf16(
                        false, af[i], false, bf_[j], (short)0, acc[i][j],
                        false, false);
        }
        __syncthreads();
    }

    // epilogue: VGPR r of tile holds (row r + 8*rsel, col l15)
#pragma unroll
    for (int i = 0; i < 2; ++i) {
        int row0 = (mtBase + wm * 2 + i) * 16 + rsel * 8;
#pragma unroll
        for (int j = 0; j < 2; ++j) {
            int col = blockIdx.x * 128 + (wn * 2 + j) * 16 + l15;
#pragma unroll
            for (int r = 0; r < 8; ++r) {
                size_t idx = (size_t)(row0 + r) * 4096 + col;
                if (ACC) C[idx] += acc[i][j][r];
                else     C[idx]  = acc[i][j][r];
            }
        }
    }
}

// ---------------------------------------------------------------------------
// Encode middle: dz = T[b]·z ; z += β dz ; pM = packed bf16( z_new ⊗ βdz )
// ---------------------------------------------------------------------------
__global__ __launch_bounds__(64) void step_mid(const float* __restrict__ T,
                                               float* __restrict__ z,
                                               unsigned short* __restrict__ pM) {
    const int b = blockIdx.x;
    const int i = threadIdx.x;
    __shared__ float zs[E_DIM], bdz[E_DIM], zn[E_DIM];

    zs[i] = z[b * E_DIM + i];
    __syncthreads();

    const float* trow = T + (size_t)b * E2_DIM + i * E_DIM;
    float acc = 0.0f;
#pragma unroll 8
    for (int j = 0; j < E_DIM; ++j) acc += trow[j] * zs[j];

    float znew = zs[i] + BETA * acc;
    zn[i] = znew;
    bdz[i] = BETA * acc;
    z[b * E_DIM + i] = znew;
    __syncthreads();

    const int mTile = b >> 4, lsub = b & 15;
#pragma unroll
    for (int u = 0; u < 4; ++u) {
        int idx = i + u * 64;                 // 0..255 -> (kc, h)
        int kc = idx >> 1, h = idx & 1;
#pragma unroll
        for (int lg = 0; lg < 2; ++lg) {
            int k0 = kc * 32 + lg * 8 + h * 16;
            u16x8 o;
#pragma unroll
            for (int e = 0; e < 8; ++e) {
                int k = k0 + e;
                o[e] = f2bf(zn[k >> 6] * bdz[k & 63]);
            }
            *(u16x8*)(pM + (((size_t)mTile * KC_CNT + kc) * 32 + lg * 16 + lsub) * 16 + h * 8) = o;
        }
    }
}

// ---------------------------------------------------------------------------
// Collapsed decode prep: dz=(lat-z0)/24; S=24*z0+276*dz; pM = packed S ⊗ βdz
// ---------------------------------------------------------------------------
__global__ __launch_bounds__(64) void decode_prep(const float* __restrict__ lat,
                                                  const float* __restrict__ z0,
                                                  unsigned short* __restrict__ pM) {
    const int b = blockIdx.x;
    const int i = threadIdx.x;
    __shared__ float bdz[E_DIM], Ss[E_DIM];

    float zi  = lat[b * E_DIM + i];
    float z0i = z0[i];
    float dzd = (zi - z0i) * (1.0f / (float)N_STEP);
    bdz[i] = BETA * dzd;
    Ss[i]  = (float)N_STEP * z0i + 276.0f * dzd;   // sum_{t=0}^{23} t = 276
    __syncthreads();

    const int mTile = b >> 4, lsub = b & 15;
#pragma unroll
    for (int u = 0; u < 4; ++u) {
        int idx = i + u * 64;
        int kc = idx >> 1, h = idx & 1;
#pragma unroll
        for (int lg = 0; lg < 2; ++lg) {
            int k0 = kc * 32 + lg * 8 + h * 16;
            u16x8 o;
#pragma unroll
            for (int e = 0; e < 8; ++e) {
                int k = k0 + e;
                o[e] = f2bf(Ss[k >> 6] * bdz[k & 63]);
            }
            *(u16x8*)(pM + (((size_t)mTile * KC_CNT + kc) * 32 + lg * 16 + lsub) * 16 + h * 8) = o;
        }
    }
}

// ---------------------------------------------------------------------------
// out[b] = mean_g (rec[b,g] - images[b,g])^2
// ---------------------------------------------------------------------------
__global__ __launch_bounds__(256) void mse_kernel(const float* __restrict__ rec,
                                                  const float* __restrict__ images,
                                                  float* __restrict__ out) {
    const int b = blockIdx.x;
    float acc = 0.0f;
    for (int g = threadIdx.x; g < G_DIM; g += 256) {
        float d = rec[(size_t)b * G_DIM + g] - images[(size_t)b * G_DIM + g];
        acc += d * d;
    }
#pragma unroll
    for (int o = 16; o > 0; o >>= 1) acc += __shfl_down(acc, o, 32);
    __shared__ float s[8];
    if ((threadIdx.x & 31) == 0) s[threadIdx.x >> 5] = acc;
    __syncthreads();
    if (threadIdx.x == 0) {
        float t = 0.0f;
#pragma unroll
        for (int w = 0; w < 8; ++w) t += s[w];
        out[b] = t * (1.0f / (float)G_DIM);
    }
}

// ---------------------------------------------------------------------------
// Host orchestration
// ---------------------------------------------------------------------------
extern "C" void kernel_launch(void* const* d_in, const int* in_sizes, int n_in,
                              void* d_out, int out_size, void* d_ws, size_t ws_size,
                              hipStream_t stream) {
    (void)in_sizes; (void)n_in; (void)out_size; (void)ws_size;

    const float* images = (const float*)d_in[0];   // (256, 4096)
    const float* xw     = (const float*)d_in[1];   // (4096, 4096)
    const float* z0     = (const float*)d_in[2];   // (64,)
    float* out = (float*)d_out;                    // (256,)

    char* ws = (char*)d_ws;
    size_t off = 0;
    auto alloc = [&](size_t bytes) -> void* {
        void* p = ws + off;
        off = (off + bytes + 255) & ~(size_t)255;
        return p;
    };
    unsigned short* pBxw  = (unsigned short*)alloc((size_t)KDIM * NDIM * 2); // packed xw
    unsigned short* pBxwT = (unsigned short*)alloc((size_t)KDIM * NDIM * 2); // packed xw^T
    unsigned short* pAx   = (unsigned short*)alloc((size_t)B_DIM * KDIM * 2); // packed (x-y)
    unsigned short* pAm   = (unsigned short*)alloc((size_t)B_DIM * KDIM * 2); // packed m / Mdec
    float*          T     = (float*)alloc((size_t)B_DIM * E2_DIM * 4);
    float*          y     = (float*)alloc((size_t)B_DIM * G_DIM * 4);
    float*          z     = (float*)alloc((size_t)B_DIM * E_DIM * 4);

    // one-time per call: weight packing + state init
    pack_B<false><<<dim3(64, 128), 256, 0, stream>>>(xw, pBxw);
    pack_B<true ><<<dim3(64, 128), 256, 0, stream>>>(xw, pBxwT);
    init_zy<<<(B_DIM * G_DIM) / 256, 256, 0, stream>>>(y, z, z0);

    const dim3 ggrid(NDIM / 128, B_DIM / 64);     // (32, 4)

    // encode: 24 sequential steps (device-wide sync via launch boundaries)
    for (int s = 0; s < N_STEP; ++s) {
        sub_conv_packed<<<512, 256, 0, stream>>>(images, y, pAx);
        gemm_wmma<false><<<ggrid, 256, 0, stream>>>(pAx, pBxw, T);
        step_mid<<<B_DIM, E_DIM, 0, stream>>>(T, z, pAm);
        gemm_wmma<true><<<ggrid, 256, 0, stream>>>(pAm, pBxwT, y);
    }

    // decode collapsed to a single GEMM (linearity of the decode recurrence)
    decode_prep<<<B_DIM, E_DIM, 0, stream>>>(z, z0, pAm);
    gemm_wmma<false><<<ggrid, 256, 0, stream>>>(pAm, pBxwT, T);
    mse_kernel<<<B_DIM, 256, 0, stream>>>(T, images, out);
}
